// DGCNN_Seg_1717986918886
// MI455X (gfx1250) — matrix operand order repaired and checked by reading, not compiled, not run on hardware
//
#include <hip/hip_runtime.h>

#define KNN 20
#define LRELU_SLOPE 0.2f
#define BN_EPS 1e-5f
#define KPTS 4   // points per KNN block

typedef __attribute__((ext_vector_type(16))) __bf16 v16bf;
typedef __attribute__((ext_vector_type(8)))  __bf16 v8bf;
typedef __attribute__((ext_vector_type(2)))  __bf16 v2bf;
typedef __attribute__((ext_vector_type(8)))  float  v8f;

union Frag16 { v16bf v; v8bf h[2]; };

// ---------------------------------------------------------------------------
// Generic GEMM: Y[b,o,n] = sum_c A(o,c) * X[b,c,n]  (+ optional per-(b,o) bias)
// A(o,c) = W[o*lda + colOff + c]                  (mode 0)
//        = W[o*lda + C + c] - W[o*lda + c]        (mode 1, edge-conv "d" term)
// bf16 A/B staged in LDS, f32 accumulate via v_wmma_f32_16x16x32_bf16.
// Block = 256 threads = 8 waves; each wave owns a 16x16 tile (block tile 32x64).
// All shape guards are uniform branches; common paths have zero predication.
// ---------------------------------------------------------------------------
__global__ __launch_bounds__(256) void gemm_wmma_kernel(
    const float* __restrict__ W, int lda, int colOff, int mode,
    const float* __restrict__ X, long xbs,
    float* __restrict__ Y, long ybs,
    const float* __restrict__ bias,
    int O, int C, int N)
{
  __shared__ __bf16 a_lds[32 * 520];   // 32 rows x (C32+8), C32 <= 512
  __shared__ __bf16 b_lds[64 * 40];    // 64 cols x 32 k, stride 40

  const int tid  = threadIdx.x;
  const int lane = tid & 31;
  const int w    = tid >> 5;
  const int b    = blockIdx.z;
  const int M0   = blockIdx.y * 32;
  const int N0   = blockIdx.x * 64;
  const int C32  = (C + 31) & ~31;
  const int CP   = C32 + 8;
  const bool cEven = (C == C32);

  // ---- stage A tile (32 x C32, zero padded) as bf16; uniform guards only ----
  for (int m = 0; m < 32; ++m) {
    const int row = M0 + m;
    __bf16* al = a_lds + m * CP;
    if (row < O) {
      const float* wr = W + (long)row * lda + colOff;
      if (mode == 0) {
        if (cEven) { for (int k = tid; k < C32; k += 256) al[k] = (__bf16)wr[k]; }
        else       { for (int k = tid; k < C32; k += 256) al[k] = (__bf16)((k < C) ? wr[k] : 0.f); }
      } else {
        if (cEven) { for (int k = tid; k < C32; k += 256) al[k] = (__bf16)(wr[C + k] - wr[k]); }
        else       { for (int k = tid; k < C32; k += 256) al[k] = (__bf16)((k < C) ? (wr[C + k] - wr[k]) : 0.f); }
      }
    } else {
      for (int k = tid; k < C32; k += 256) al[k] = (__bf16)0.f;
    }
  }

  const float* Xb = X + (long)b * xbs;

  // per-wave fragment addressing (ISA 7.12.2 layouts)
  const int mloc   = (w >> 2) * 16 + (lane & 15);
  const int khalf  = lane >> 4;
  const __bf16* aBase = a_lds + mloc * CP + khalf * 8;   // + k0, + k0+16
  const int nloc   = (w & 3) * 16 + (lane & 15);
  const __bf16* bBase = b_lds + nloc * 40 + khalf * 8;   // + 0, + 16
  const int ncol   = N0 + nloc;

  v8f acc = {};
  for (int k0 = 0; k0 < C32; k0 += 32) {
    __syncthreads();   // prior-iteration fragment reads (and A staging) done
    // ---- stage B tile (32 k x 64 n) as bf16, coalesced 64-wide loads ----
    if (k0 + 32 <= C) {            // uniform fast path: no guards
#pragma unroll
      for (int t = 0; t < 4; ++t) {
        int i  = tid + t * 256;
        int nl = i & 63;
        int kp = (i >> 6) * 2;
        int off = (k0 + kp) * N + N0 + nl;     // fits in int (C*N <= 1M)
        float v0 = Xb[off];
        float v1 = Xb[off + N];
        v2bf p; p.x = (__bf16)v0; p.y = (__bf16)v1;
        *(v2bf*)&b_lds[nl * 40 + kp] = p;
      }
    } else {                        // tail (only the C=3 layer hits this)
#pragma unroll
      for (int t = 0; t < 4; ++t) {
        int i  = tid + t * 256;
        int nl = i & 63;
        int kp = (i >> 6) * 2;
        int kk = k0 + kp;
        int off = kk * N + N0 + nl;
        float v0 = (kk     < C) ? Xb[off]     : 0.f;
        float v1 = (kk + 1 < C) ? Xb[off + N] : 0.f;
        v2bf p; p.x = (__bf16)v0; p.y = (__bf16)v1;
        *(v2bf*)&b_lds[nl * 40 + kp] = p;
      }
    }
    __syncthreads();

    Frag16 af, bf;
    af.h[0] = *(const v8bf*)(aBase + k0);
    af.h[1] = *(const v8bf*)(aBase + k0 + 16);
    bf.h[0] = *(const v8bf*)(bBase);
    bf.h[1] = *(const v8bf*)(bBase + 16);
    acc = __builtin_amdgcn_wmma_f32_16x16x32_bf16(
        false, af.v, false, bf.v, (short)0, acc, false, false);
  }

  // C/D layout: VGPR r -> M = r + 8*(lane>=16), N = lane&15
  const int rbase = M0 + ((lane >> 4) ? 8 : 0) + ((w >> 2) * 16);
#pragma unroll
  for (int r = 0; r < 8; ++r) {
    int row = rbase + r;
    if (row < O) {
      float v = acc[r];
      if (bias) v += bias[(long)b * O + row];
      Y[(long)b * ybs + (long)row * N + ncol] = v;
    }
  }
}

// ---------------------------------------------------------------------------
// KNN: 4 points per block (128 threads). Distances for all 2048 candidates
// kept in LDS; each x_m column loaded once and reused for 4 centers.
// Top-20 per point done by one wave with shfl_xor argmax (ties -> lower index,
// matching jax.lax.top_k). Self-distance included, as in the reference.
// ---------------------------------------------------------------------------
__global__ __launch_bounds__(128) void knn_kernel(
    const float* __restrict__ X, long xbs, int C, int N, int* __restrict__ idx)
{
  __shared__ float ctr[KPTS * 128];
  __shared__ float xxn[KPTS];
  __shared__ float dist[KPTS * 2048];
  const int tid = threadIdx.x;
  const int n0 = blockIdx.x * KPTS;
  const int b  = blockIdx.y;
  const float* Xb = X + (long)b * xbs;

  for (int i = tid; i < KPTS * C; i += 128) {
    int p = i / C, c = i - p * C;
    ctr[p * 128 + c] = Xb[(long)c * N + (n0 + p)];
  }
  __syncthreads();
  if (tid < KPTS) {
    float s = 0.f;
    for (int c = 0; c < C; ++c) { float v = ctr[tid * 128 + c]; s += v * v; }
    xxn[tid] = s;
  }
  __syncthreads();

  for (int m = tid; m < N; m += 128) {
    float dot[KPTS];
#pragma unroll
    for (int p = 0; p < KPTS; ++p) dot[p] = 0.f;
    float xxm = 0.f;
    for (int c = 0; c < C; ++c) {
      float xm = Xb[(long)c * N + m];
      xxm += xm * xm;
#pragma unroll
      for (int p = 0; p < KPTS; ++p) dot[p] += xm * ctr[p * 128 + c];
    }
#pragma unroll
    for (int p = 0; p < KPTS; ++p)
      dist[p * 2048 + m] = 2.f * dot[p] - xxm - xxn[p];
  }
  __syncthreads();

  // wave p selects top-KNN for point n0+p
  const int p = tid >> 5;
  const int lane = tid & 31;
  float* dw = dist + p * 2048;
  int* out = idx + ((long)b * N + (n0 + p)) * KNN;
  for (int j = 0; j < KNN; ++j) {
    float bv = -3.4e38f; int bi = 0;
    for (int m = lane; m < N; m += 32) {
      float v = dw[m];
      if (v > bv || (v == bv && m < bi)) { bv = v; bi = m; }
    }
#pragma unroll
    for (int s = 16; s > 0; s >>= 1) {
      float ov = __shfl_xor(bv, s, 32);
      int   oi = __shfl_xor(bi, s, 32);
      if (ov > bv || (ov == bv && oi < bi)) { bv = ov; bi = oi; }
    }
    if (lane == 0) out[j] = bi;
    dw[bi] = -3.4e38f;   // all lanes write same value
  }
}

// ---------------------------------------------------------------------------
// Edge-conv BN stats over (b,n,k) of h = y1[b,o,idx(b,n,k)] + d[b,o,n].
// Nested loops: no integer division in the hot path.
// ---------------------------------------------------------------------------
__global__ __launch_bounds__(256) void edge_stats_kernel(
    const float* __restrict__ y1, const float* __restrict__ d,
    const int* __restrict__ idx, int O, int N, int B,
    float* __restrict__ mean, float* __restrict__ var)
{
  __shared__ float ssum[256], ssq[256];
  const int o = blockIdx.x, tid = threadIdx.x;
  float s = 0.f, q = 0.f;
  for (int b2 = 0; b2 < B; ++b2) {
    const float* y1r = y1 + ((long)b2 * O + o) * N;
    const float* dr  = d  + ((long)b2 * O + o) * N;
    const int*   ib  = idx + (long)b2 * N * KNN;
    for (int n = tid; n < N; n += 256) {
      float dd = dr[n];
      const int* ir = ib + n * KNN;
      for (int k = 0; k < KNN; ++k) {
        float h = y1r[ir[k]] + dd;
        s += h; q += h * h;
      }
    }
  }
  ssum[tid] = s; ssq[tid] = q;
  __syncthreads();
  for (int st = 128; st > 0; st >>= 1) {
    if (tid < st) { ssum[tid] += ssum[tid + st]; ssq[tid] += ssq[tid + st]; }
    __syncthreads();
  }
  if (tid == 0) {
    const float total = (float)B * (float)N * (float)KNN;
    float m = ssum[0] / total;
    mean[o] = m;
    var[o]  = ssq[0] / total - m * m;
  }
}

// ---------------------------------------------------------------------------
// Edge-conv epilogue: out[b,o,n] = max_k lrelu(bn(h)), written into xcat slice.
// ---------------------------------------------------------------------------
__global__ __launch_bounds__(256) void edge_maxk_kernel(
    const float* __restrict__ y1, const float* __restrict__ d,
    const int* __restrict__ idx,
    const float* __restrict__ mean, const float* __restrict__ var,
    const float* __restrict__ gam, const float* __restrict__ bet,
    int O, int N, float* __restrict__ out, long obs, long ooff)
{
  const int n = blockIdx.x * 256 + threadIdx.x;
  const int o = blockIdx.y, b = blockIdx.z;
  float a = gam[o] * rsqrtf(var[o] + BN_EPS);
  float c = bet[o] - mean[o] * a;
  const float* y1r = y1 + ((long)b * O + o) * N;
  float dd = d[((long)b * O + o) * N + n];
  const int* ir = idx + ((long)b * N + n) * KNN;
  float m = -3.4e38f;
  for (int k = 0; k < KNN; ++k) {
    float h = y1r[ir[k]] + dd;
    float t = a * h + c;
    t = (t >= 0.f) ? t : LRELU_SLOPE * t;
    m = fmaxf(m, t);
  }
  out[(long)b * obs + ooff + (long)o * N + n] = m;
}

// ---------------------------------------------------------------------------
// Dense BN stats over (b,n) for [B,C,N] buffers; one block per channel.
// ---------------------------------------------------------------------------
__global__ __launch_bounds__(256) void bn_stats_kernel(
    const float* __restrict__ buf, int C, int N, int B,
    float* __restrict__ mean, float* __restrict__ var)
{
  __shared__ float ssum[256], ssq[256];
  const int c = blockIdx.x, tid = threadIdx.x;
  float s = 0.f, q = 0.f;
  for (int b2 = 0; b2 < B; ++b2) {
    const float* r = buf + ((long)b2 * C + c) * N;
    for (int n = tid; n < N; n += 256) {
      float v = r[n];
      s += v; q += v * v;
    }
  }
  ssum[tid] = s; ssq[tid] = q;
  __syncthreads();
  for (int st = 128; st > 0; st >>= 1) {
    if (tid < st) { ssum[tid] += ssum[tid + st]; ssq[tid] += ssq[tid + st]; }
    __syncthreads();
  }
  if (tid == 0) {
    const float total = (float)B * (float)N;
    float m = ssum[0] / total;
    mean[c] = m;
    var[c]  = ssq[0] / total - m * m;
  }
}

// grid (N/256, C, B): no index division.
__global__ __launch_bounds__(256) void bn_lrelu_kernel(
    float* __restrict__ buf, const float* __restrict__ mean,
    const float* __restrict__ var, const float* __restrict__ gam,
    const float* __restrict__ bet, int C, int N)
{
  const int n = blockIdx.x * 256 + threadIdx.x;
  const int c = blockIdx.y, b = blockIdx.z;
  float a = gam[c] * rsqrtf(var[c] + BN_EPS);
  float cc = bet[c] - mean[c] * a;
  long i = ((long)b * C + c) * N + n;
  float t = a * buf[i] + cc;
  buf[i] = (t >= 0.f) ? t : LRELU_SLOPE * t;
}

__global__ __launch_bounds__(256) void rowmax_kernel(
    const float* __restrict__ buf, int C, int N, float* __restrict__ out)
{
  __shared__ float sm[256];
  const int c = blockIdx.x, b = blockIdx.y, tid = threadIdx.x;
  const float* r = buf + ((long)b * C + c) * N;
  float m = -3.4e38f;
  for (int n = tid; n < N; n += 256) m = fmaxf(m, r[n]);
  sm[tid] = m; __syncthreads();
  for (int s = 128; s > 0; s >>= 1) {
    if (tid < s) sm[tid] = fmaxf(sm[tid], sm[tid + s]);
    __syncthreads();
  }
  if (tid == 0) out[(long)b * C + c] = sm[0];
}

// Label branch: lf[b,o] = lrelu(bn_over_batch(W206 @ l)), 64 channels, B=8.
__global__ void lf_kernel(const float* __restrict__ l, const float* __restrict__ W,
                          const float* __restrict__ gam, const float* __restrict__ bet,
                          float* __restrict__ lf, int B)
{
  int o = threadIdx.x;  // 64 threads
  float v[8];
  float s = 0.f, q = 0.f;
  for (int b = 0; b < B; ++b) {
    float acc = 0.f;
    for (int j = 0; j < 16; ++j) acc += W[o * 16 + j] * l[b * 16 + j];
    v[b] = acc; s += acc; q += acc * acc;
  }
  float m = s / (float)B, vr = q / (float)B - m * m;
  float a = gam[o] * rsqrtf(vr + BN_EPS);
  float c = bet[o] - m * a;
  for (int b = 0; b < B; ++b) {
    float t = a * v[b] + c;
    lf[b * 64 + o] = (t >= 0.f) ? t : LRELU_SLOPE * t;
  }
}

// Broadcast-global term of W207: glob is constant over n, so its contribution
// collapses to a per-(b,o) bias c207[b,o] = W207[:, :1024]@gmax + W207[:,1024:1088]@lf.
__global__ void c207_kernel(const float* __restrict__ W207, const float* __restrict__ gmax,
                            const float* __restrict__ lf, float* __restrict__ c207)
{
  int o = threadIdx.x;  // 256
  int b = blockIdx.x;   // B
  const float* wr = W207 + (long)o * 1600;
  float s = 0.f;
  for (int j = 0; j < 1024; ++j) s += wr[j] * gmax[b * 1024 + j];
  for (int j = 0; j < 64; ++j)  s += wr[1024 + j] * lf[b * 64 + j];
  c207[b * 256 + o] = s;
}

// ---------------------------------------------------------------------------
extern "C" void kernel_launch(void* const* d_in, const int* in_sizes, int n_in,
                              void* d_out, int out_size, void* d_ws, size_t ws_size,
                              hipStream_t stream)
{
  const int B = 8, N = 2048;
  const float* x    = (const float*)d_in[0];
  const float* l    = (const float*)d_in[1];
  const float* W1   = (const float*)d_in[2];
  const float* g1   = (const float*)d_in[3];
  const float* b1   = (const float*)d_in[4];
  const float* W2   = (const float*)d_in[5];
  const float* g2   = (const float*)d_in[6];
  const float* b2   = (const float*)d_in[7];
  const float* W3   = (const float*)d_in[8];
  const float* g3   = (const float*)d_in[9];
  const float* b3   = (const float*)d_in[10];
  const float* W4   = (const float*)d_in[11];
  const float* g4   = (const float*)d_in[12];
  const float* b4   = (const float*)d_in[13];
  const float* W5   = (const float*)d_in[14];
  const float* g5   = (const float*)d_in[15];
  const float* b5   = (const float*)d_in[16];
  const float* W206 = (const float*)d_in[17];
  const float* g206 = (const float*)d_in[18];
  const float* b206 = (const float*)d_in[19];
  const float* W207 = (const float*)d_in[20];
  const float* g207 = (const float*)d_in[21];
  const float* b207 = (const float*)d_in[22];
  const float* W208 = (const float*)d_in[23];
  const float* g208 = (const float*)d_in[24];
  const float* b208 = (const float*)d_in[25];
  const float* W209 = (const float*)d_in[26];
  const float* g209 = (const float*)d_in[27];
  const float* b209 = (const float*)d_in[28];
  const float* W2010= (const float*)d_in[29];
  float* out = (float*)d_out;

  char* ws = (char*)d_ws;
  size_t off = 0;
  auto alloc = [&](size_t bytes) -> void* {
    void* p = ws + off;
    off += (bytes + 255) & ~(size_t)255;
    return p;
  };
  int*   idx   = (int*)  alloc((size_t)B * N * KNN * 4);
  float* y1    = (float*)alloc((size_t)B * 256 * N * 4);
  float* dbuf  = (float*)alloc((size_t)B * 256 * N * 4);
  float* xcat  = (float*)alloc((size_t)B * 512 * N * 4);   // x1|x2|x3|x4 channel slices
  float* h5    = (float*)alloc((size_t)B * 1024 * N * 4);
  float* h207  = (float*)alloc((size_t)B * 256 * N * 4);
  float* h208  = (float*)alloc((size_t)B * 256 * N * 4);
  float* h209  = (float*)alloc((size_t)B * 128 * N * 4);
  float* gmaxb = (float*)alloc((size_t)B * 1024 * 4);
  float* lfb   = (float*)alloc((size_t)B * 64 * 4);
  float* c207b = (float*)alloc((size_t)B * 256 * 4);
  float* meanb = (float*)alloc(1024 * 4);
  float* varb  = (float*)alloc(1024 * 4);

  dim3 blk(256);
  auto gemm = [&](const float* W, int lda, int colOff, int mode,
                  const float* X, long xbs, float* Y, long ybs,
                  const float* bias, int O, int C) {
    dim3 grid(N / 64, (O + 31) / 32, B);
    gemm_wmma_kernel<<<grid, blk, 0, stream>>>(W, lda, colOff, mode, X, xbs, Y, ybs,
                                               bias, O, C, N);
  };
  auto bnact = [&](float* buf, const float* g, const float* bb, int C) {
    bn_stats_kernel<<<C, blk, 0, stream>>>(buf, C, N, B, meanb, varb);
    bn_lrelu_kernel<<<dim3(N / 256, C, B), blk, 0, stream>>>(buf, meanb, varb, g, bb, C, N);
  };

  lf_kernel<<<1, 64, 0, stream>>>(l, W206, g206, b206, lfb, B);

  auto edge = [&](const float* Xin, long xbs, int C, const float* W,
                  const float* g, const float* bb, int O, long chanOff) {
    knn_kernel<<<dim3(N / KPTS, B), dim3(128), 0, stream>>>(Xin, xbs, C, N, idx);
    gemm(W, 2 * C, 0, 0, Xin, xbs, y1,   (long)O * N, nullptr, O, C);   // W[:, :C]@x
    gemm(W, 2 * C, 0, 1, Xin, xbs, dbuf, (long)O * N, nullptr, O, C);   // (W[:,C:]-W[:,:C])@x
    edge_stats_kernel<<<O, blk, 0, stream>>>(y1, dbuf, idx, O, N, B, meanb, varb);
    edge_maxk_kernel<<<dim3(N / 256, O, B), blk, 0, stream>>>(
        y1, dbuf, idx, meanb, varb, g, bb, O, N, xcat, 512L * N, chanOff * N);
  };

  edge(x,               3L * N,   3,   W1, g1, b1, 64,  0);
  edge(xcat,            512L * N, 64,  W2, g2, b2, 64,  64);
  edge(xcat + 64 * N,   512L * N, 64,  W3, g3, b3, 128, 128);
  edge(xcat + 128 * N,  512L * N, 128, W4, g4, b4, 256, 256);

  // conv5: [1024,512] @ xcat
  gemm(W5, 512, 0, 0, xcat, 512L * N, h5, 1024L * N, nullptr, 1024, 512);
  bnact(h5, g5, b5, 1024);
  rowmax_kernel<<<dim3(1024, B), blk, 0, stream>>>(h5, 1024, N, gmaxb);

  c207_kernel<<<B, 256, 0, stream>>>(W207, gmaxb, lfb, c207b);

  // conv207: global part as bias + W207[:,1088:1600] @ xcat
  gemm(W207, 1600, 1088, 0, xcat, 512L * N, h207, 256L * N, c207b, 256, 512);
  bnact(h207, g207, b207, 256);

  gemm(W208, 256, 0, 0, h207, 256L * N, h208, 256L * N, nullptr, 256, 256);
  bnact(h208, g208, b208, 256);

  gemm(W209, 256, 0, 0, h208, 256L * N, h209, 128L * N, nullptr, 128, 256);
  bnact(h209, g209, b209, 128);

  // final head: [50,128] @ h209 -> d_out [B,50,N]
  gemm(W2010, 128, 0, 0, h209, 128L * N, out, 50L * N, nullptr, 50, 128);
}
